// MuleHunterGNN_8117488189976
// MI455X (gfx1250) — matrix operand order repaired and checked by reading, not compiled, not run on hardware
//
#include <hip/hip_runtime.h>
#include <hip/hip_bf16.h>
#include <stdint.h>

typedef __bf16 bf16;
typedef __attribute__((ext_vector_type(16))) __bf16 v16bf;
typedef __attribute__((ext_vector_type(8)))  __bf16 v8bf;
typedef __attribute__((ext_vector_type(8)))  float  v8f;

#define NN 100000L
#define EE 400000L
#define EPS 1e-5f

// ---------- helpers ----------
__device__ __forceinline__ unsigned f2ord(float f) {
  unsigned u = __float_as_uint(f);
  return (u & 0x80000000u) ? ~u : (u | 0x80000000u);
}
__device__ __forceinline__ float ord2f(unsigned u) {
  return (u & 0x80000000u) ? __uint_as_float(u & 0x7fffffffu) : __uint_as_float(~u);
}

// ---------- elementwise / prep ----------
__global__ void k_zero_f32(float* p, long n) {
  long i = (long)blockIdx.x * blockDim.x + threadIdx.x;
  if (i < n) p[i] = 0.f;
}
__global__ void k_fill_u32(unsigned* p, unsigned v, long n) {
  long i = (long)blockIdx.x * blockDim.x + threadIdx.x;
  if (i < n) p[i] = v;
}
__global__ void k_f32_to_bf16(bf16* d, const float* s, long n) {
  long i = (long)blockIdx.x * blockDim.x + threadIdx.x;
  if (i < n) d[i] = (bf16)s[i];
}
// dst[m*K + k] = src[k*M + m]   (W[K][M] -> Wt[M][K], f32 -> bf16)
__global__ void k_transpose_bf16(bf16* d, const float* s, int K, int M) {
  long i = (long)blockIdx.x * blockDim.x + threadIdx.x;
  if (i < (long)K * M) {
    int k = (int)(i / M), m = (int)(i % M);
    d[(long)m * K + k] = (bf16)s[i];
  }
}

// ---------- graph scatter ----------
__global__ void k_deg(float* deg, const int* dst, long E) {
  long i = (long)blockIdx.x * blockDim.x + threadIdx.x;
  if (i < E) atomicAdd(&deg[dst[i]], 1.f);
}
__global__ void k_scatter_add_f32(float* out, const float* X, const int* src,
                                  const int* dst, long total, int C) {
  long i = (long)blockIdx.x * blockDim.x + threadIdx.x;
  if (i < total) {
    long e = i / C; int c = (int)(i % C);
    int s = src[e], d = dst[e];
    atomicAdd(&out[(long)d * C + c], X[(long)s * C + c]);
  }
}
__global__ void k_scatter_add_bf16(float* out, const bf16* X, const int* src,
                                   const int* dst, long total, int C) {
  long i = (long)blockIdx.x * blockDim.x + threadIdx.x;
  if (i < total) {
    long e = i / C; int c = (int)(i % C);
    int s = src[e], d = dst[e];
    atomicAdd(&out[(long)d * C + c], (float)X[(long)s * C + c]);
  }
}
__global__ void k_mean_bf16(bf16* m, const float* agg, const float* deg, long total, int C) {
  long i = (long)blockIdx.x * blockDim.x + threadIdx.x;
  if (i < total) {
    long r = i / C;
    m[i] = (bf16)(agg[i] / fmaxf(deg[r], 1.f));
  }
}

// ---------- bf16 WMMA GEMM:  C[rows x Kout] = A0@W0 (+ A1@W1) + bias ----------
// A: row-major [rows x Kdim] bf16.  W*: transposed weights Wt[Kout][Kdim] bf16.
// grid.x = rows/16, blockDim = 256 (8 waves).
// The shared 16-row A tile(s) are staged once per block into LDS with the
// gfx1250 async-to-LDS path (ASYNCcnt), then each wave computes PAIRS of
// 16x16 output tiles so one A fragment feeds 2 (dual: 4) WMMAs.
__global__ void k_gemm(const bf16* __restrict__ A0, const bf16* __restrict__ A1,
                       const bf16* __restrict__ W0, const bf16* __restrict__ W1,
                       const float* __restrict__ bias,
                       float* __restrict__ Cf, bf16* __restrict__ Cb,
                       int Kdim, int Kout, int relu) {
  __shared__ __attribute__((aligned(16))) bf16 sA0[16 * 128];
  __shared__ __attribute__((aligned(16))) bf16 sA1[16 * 128];
  const int tid  = threadIdx.x;
  const int lane = tid & 31;
  const int wv   = tid >> 5;
  const int half = lane >> 4;
  const int mloc = lane & 15;
  const long rowbase = (long)blockIdx.x * 16;
  const int nct = Kout >> 4;

  // ---- cooperative async stage of the 16xKdim A tile(s) into LDS ----
  const int chunks = Kdim * 2;                 // 16-byte chunks per tile
  if (tid < chunks) {
    unsigned l0 = (unsigned)(uintptr_t)(void*)sA0 + tid * 16;
    unsigned long long g0 =
        (unsigned long long)(uintptr_t)((const char*)(A0 + rowbase * Kdim) + tid * 16);
    asm volatile("global_load_async_to_lds_b128 %0, %1, off"
                 :: "v"(l0), "v"(g0) : "memory");
    if (A1) {
      unsigned l1 = (unsigned)(uintptr_t)(void*)sA1 + tid * 16;
      unsigned long long g1 =
          (unsigned long long)(uintptr_t)((const char*)(A1 + rowbase * Kdim) + tid * 16);
      asm volatile("global_load_async_to_lds_b128 %0, %1, off"
                   :: "v"(l1), "v"(g1) : "memory");
    }
  }
  asm volatile("s_wait_asynccnt 0" ::: "memory");
  __syncthreads();

  for (int ct = wv * 2; ct < nct; ct += 16) {  // nct is even for all our shapes
    const int ncol0 = (ct << 4) + mloc;
    const int ncol1 = ncol0 + 16;
    float bi0 = bias ? bias[ncol0] : 0.f;
    float bi1 = bias ? bias[ncol1] : 0.f;
    v8f acc0, acc1;
#pragma unroll
    for (int i = 0; i < 8; i++) { acc0[i] = bi0; acc1[i] = bi1; }
    for (int kc = 0; kc < Kdim; kc += 32) {
      // A fragment (from LDS): elems 0..7 -> k=kc+half*8+{0..7}; 8..15 -> +16
      const bf16* pa = sA0 + mloc * Kdim + kc + half * 8;
      v8bf a_lo = *(const v8bf*)pa;
      v8bf a_hi = *(const v8bf*)(pa + 16);
      // B fragments: elems 0..7 -> k=kc+half*8+{0..7}; 8..15 -> +16, at fixed output col
      const bf16* pb0 = W0 + (long)ncol0 * Kdim + kc + half * 8;
      const bf16* pb1 = W0 + (long)ncol1 * Kdim + kc + half * 8;
      v8bf b0_lo = *(const v8bf*)pb0;
      v8bf b0_hi = *(const v8bf*)(pb0 + 16);
      v8bf b1_lo = *(const v8bf*)pb1;
      v8bf b1_hi = *(const v8bf*)(pb1 + 16);
      v16bf a, b0, b1;
#pragma unroll
      for (int i = 0; i < 8; i++) {
        a[i] = a_lo[i];  a[8 + i] = a_hi[i];
        b0[i] = b0_lo[i]; b0[8 + i] = b0_hi[i];
        b1[i] = b1_lo[i]; b1[8 + i] = b1_hi[i];
      }
      acc0 = __builtin_amdgcn_wmma_f32_16x16x32_bf16(false, a, false, b0,
                                                     (short)0, acc0, false, false);
      acc1 = __builtin_amdgcn_wmma_f32_16x16x32_bf16(false, a, false, b1,
                                                     (short)0, acc1, false, false);
      if (A1) {
        const bf16* pa1 = sA1 + mloc * Kdim + kc + half * 8;
        v8bf c_lo = *(const v8bf*)pa1;
        v8bf c_hi = *(const v8bf*)(pa1 + 16);
        const bf16* pc0 = W1 + (long)ncol0 * Kdim + kc + half * 8;
        const bf16* pc1 = W1 + (long)ncol1 * Kdim + kc + half * 8;
        v8bf d0_lo = *(const v8bf*)pc0;
        v8bf d0_hi = *(const v8bf*)(pc0 + 16);
        v8bf d1_lo = *(const v8bf*)pc1;
        v8bf d1_hi = *(const v8bf*)(pc1 + 16);
        v16bf a2, c0, c1;
#pragma unroll
        for (int i = 0; i < 8; i++) {
          a2[i] = c_lo[i];  a2[8 + i] = c_hi[i];
          c0[i] = d0_lo[i]; c0[8 + i] = d0_hi[i];
          c1[i] = d1_lo[i]; c1[8 + i] = d1_hi[i];
        }
        acc0 = __builtin_amdgcn_wmma_f32_16x16x32_bf16(false, a2, false, c0,
                                                       (short)0, acc0, false, false);
        acc1 = __builtin_amdgcn_wmma_f32_16x16x32_bf16(false, a2, false, c1,
                                                       (short)0, acc1, false, false);
      }
    }
#pragma unroll
    for (int v = 0; v < 8; v++) {
      long row = rowbase + half * 8 + v;           // C layout: m = half*8 + v
      float v0 = acc0[v], v1 = acc1[v];
      if (relu) { v0 = fmaxf(v0, 0.f); v1 = fmaxf(v1, 0.f); }
      long o0 = row * (long)Kout + ncol0;
      long o1 = row * (long)Kout + ncol1;
      if (Cf) { Cf[o0] = v0; Cf[o1] = v1; }
      if (Cb) { Cb[o0] = (bf16)v0; Cb[o1] = (bf16)v1; }
    }
  }
}

// ---------- BatchNorm: column sums/sumsq reduction (LDS) + apply ----------
__global__ void k_colstats(const float* X, float* st, long n_rows, int C) {
  __shared__ float sh1[256], sh2[256];
  int tid = threadIdx.x;
  int c = tid % C, grp = tid / C, G = blockDim.x / C;
  float s = 0.f, q = 0.f;
  for (long r = (long)blockIdx.x * G + grp; r < n_rows; r += (long)gridDim.x * G) {
    float v = X[r * C + c];
    s += v; q += v * v;
  }
  sh1[tid] = s; sh2[tid] = q;
  __syncthreads();
  if (grp == 0) {
    for (int g = 1; g < G; g++) { s += sh1[g * C + c]; q += sh2[g * C + c]; }
    atomicAdd(&st[c], s);
    atomicAdd(&st[C + c], q);
  }
}
__global__ void k_bn_apply(const float* X, const float* st, const float* g,
                           const float* b, float* Yf, bf16* Yb,
                           long total, int C, float inv_n, int relu) {
  long i = (long)blockIdx.x * blockDim.x + threadIdx.x;
  if (i < total) {
    int c = (int)(i % C);
    float mean = st[c] * inv_n;
    float var = st[C + c] * inv_n - mean * mean;
    float y = g[c] * (X[i] - mean) * rsqrtf(var + EPS) + b[c];
    if (relu) y = fmaxf(y, 0.f);
    if (Yf) Yf[i] = y;
    if (Yb) Yb[i] = (bf16)y;
  }
}

// ---------- GAT ----------
__global__ void k_attn_scores(float* a_s, float* a_d, const bf16* xh,
                              const float* atts, const float* attd, long total) {
  long i = (long)blockIdx.x * blockDim.x + threadIdx.x;
  if (i < total) {
    long node = i >> 2; int h = (int)(i & 3);
    const bf16* row = xh + node * 512 + h * 128;
    const float* ws = atts + h * 128;
    const float* wd = attd + h * 128;
    float s = 0.f, d = 0.f;
#pragma unroll 4
    for (int c = 0; c < 128; c++) {
      float v = (float)row[c];
      s += v * ws[c]; d += v * wd[c];
    }
    a_s[i] = s; a_d[i] = d;
  }
}
__global__ void k_edge_scores(float* ebuf, unsigned* emax, const float* a_s,
                              const float* a_d, const int* src, const int* dst, long total) {
  long i = (long)blockIdx.x * blockDim.x + threadIdx.x;
  if (i < total) {
    long e = i >> 2; int h = (int)(i & 3);
    int s = src[e], d = dst[e];
    float v = a_s[(long)s * 4 + h] + a_d[(long)d * 4 + h];
    v = (v > 0.f) ? v : 0.2f * v;        // leaky_relu 0.2
    ebuf[i] = v;
    atomicMax(&emax[(long)d * 4 + h], f2ord(v));
  }
}
__global__ void k_edge_exp(float* ebuf, float* den, const unsigned* emax,
                           const int* dst, long total) {
  long i = (long)blockIdx.x * blockDim.x + threadIdx.x;
  if (i < total) {
    long e = i >> 2; int h = (int)(i & 3);
    int d = dst[e];
    float ee = expf(ebuf[i] - ord2f(emax[(long)d * 4 + h]));
    ebuf[i] = ee;
    atomicAdd(&den[(long)d * 4 + h], ee);
  }
}
// out[dst,c] += sum_h alpha[e,h] * xh[src, h*128+c]
__global__ void k_gat_agg(float* gout, const bf16* xh, const float* ebuf,
                          const float* den, const int* src, const int* dst, long total) {
  long i = (long)blockIdx.x * blockDim.x + threadIdx.x;
  if (i < total) {
    long e = i >> 7; int c = (int)(i & 127);
    int s = src[e], d = dst[e];
    const bf16* xr = xh + (long)s * 512 + c;
    __builtin_prefetch(xr, 0, 0);
    float acc = 0.f;
#pragma unroll
    for (int h = 0; h < 4; h++) {
      float alpha = ebuf[e * 4 + h] / fmaxf(den[(long)d * 4 + h], 1e-16f);
      acc += alpha * (float)xr[h * 128];
    }
    atomicAdd(&gout[(long)d * 128 + c], acc);
  }
}
// X = a*X + bias[col]
__global__ void k_axpb(float* X, float a, const float* bias, long total, int C) {
  long i = (long)blockIdx.x * blockDim.x + threadIdx.x;
  if (i < total) X[i] = a * X[i] + bias[i % C];
}
__global__ void k_add_emb(bf16* emb, const float* h3, const float* iden, long n) {
  long i = (long)blockIdx.x * blockDim.x + threadIdx.x;
  if (i < n) emb[i] = (bf16)(h3[i] + iden[i]);
}

// ---------- classifier head: logits (32->2) + log_softmax ----------
__global__ void k_head(float* out, const bf16* z2, const float* Wc3,
                       const float* bc3, long n) {
  long i = (long)blockIdx.x * blockDim.x + threadIdx.x;
  if (i < n) {
    const bf16* zr = z2 + i * 32;
    float l0 = bc3[0], l1 = bc3[1];
#pragma unroll
    for (int c = 0; c < 32; c++) {
      float v = (float)zr[c];
      l0 += v * Wc3[c * 2 + 0];
      l1 += v * Wc3[c * 2 + 1];
    }
    float mx = fmaxf(l0, l1);
    float lse = mx + logf(expf(l0 - mx) + expf(l1 - mx));
    out[i * 2 + 0] = l0 - lse;
    out[i * 2 + 1] = l1 - lse;
  }
}

// ================= host side =================
extern "C" void kernel_launch(void* const* d_in, const int* in_sizes, int n_in,
                              void* d_out, int out_size, void* d_ws, size_t ws_size,
                              hipStream_t stream) {
  (void)in_sizes; (void)n_in; (void)out_size; (void)ws_size;
  const float* x    = (const float*)d_in[0];
  const int*   ei   = (const int*)d_in[1];
  const int*   srcI = ei;
  const int*   dstI = ei + EE;
  const float* Wl1 = (const float*)d_in[2];
  const float* Wr1 = (const float*)d_in[3];
  const float* b1  = (const float*)d_in[4];
  const float* g1  = (const float*)d_in[5];
  const float* be1 = (const float*)d_in[6];
  const float* Wg  = (const float*)d_in[7];
  const float* atts= (const float*)d_in[8];
  const float* attd= (const float*)d_in[9];
  const float* bg  = (const float*)d_in[10];
  const float* g2  = (const float*)d_in[11];
  const float* be2 = (const float*)d_in[12];
  const float* Wl3 = (const float*)d_in[13];
  const float* Wr3 = (const float*)d_in[14];
  const float* b3  = (const float*)d_in[15];
  const float* g3  = (const float*)d_in[16];
  const float* be3 = (const float*)d_in[17];
  const float* Wsk = (const float*)d_in[18];
  const float* bsk = (const float*)d_in[19];
  const float* Wc1 = (const float*)d_in[20];
  const float* bc1 = (const float*)d_in[21];
  const float* gc  = (const float*)d_in[22];
  const float* bec = (const float*)d_in[23];
  const float* Wc2 = (const float*)d_in[24];
  const float* bc2 = (const float*)d_in[25];
  const float* Wc3 = (const float*)d_in[26];
  const float* bc3 = (const float*)d_in[27];
  float* out = (float*)d_out;

  // ---- workspace bump allocator with deliberate lifetime reuse ----
  char* ws = (char*)d_ws;
  size_t off = 0;
  auto al = [&](size_t b) -> char* {
    char* r = ws + off;
    off = (off + b + 255) & ~(size_t)255;
    return r;
  };
  bf16* X16  = (bf16*)al(NN * 128 * 2);   // x in bf16 (persist)
  bf16* WL1T = (bf16*)al(128 * 128 * 2);
  bf16* WR1T = (bf16*)al(128 * 128 * 2);
  bf16* WGT  = (bf16*)al(512 * 128 * 2);
  bf16* WL3T = (bf16*)al(64 * 128 * 2);
  bf16* WR3T = (bf16*)al(64 * 128 * 2);
  bf16* WSKT = (bf16*)al(64 * 128 * 2);
  bf16* WC1T = (bf16*)al(64 * 64 * 2);
  bf16* WC2T = (bf16*)al(32 * 64 * 2);
  float* A32 = (float*)al(NN * 128 * 4);  // reuse: agg1 -> h1_pre -> gout -> agg3
  float* B32 = (float*)al(NN * 64 * 4);   // reuse: h3_pre -> z1_pre
  char*  M16r= al(NN * 128 * 2);          // reuse: mean1/mean3 bf16 -> h3 f32 -> z2 bf16
  bf16*  M16 = (bf16*)M16r;
  float* H3F = (float*)M16r;
  bf16*  Z2B = (bf16*)M16r;
  char*  H1r = al(NN * 128 * 2);          // reuse: h1 bf16 -> [emb | z1] bf16
  bf16*  H1B = (bf16*)H1r;
  bf16*  EMB = (bf16*)H1r;
  bf16*  Z1B = (bf16*)(H1r + NN * 64 * 2);
  bf16*  H2B = (bf16*)al(NN * 128 * 2);   // h2 bf16
  bf16*  XH  = (bf16*)al(NN * 512 * 2);   // GAT projection, bf16
  float* DEG = (float*)al(NN * 4);
  float* AS  = (float*)al(NN * 4 * 4);
  float* AD  = (float*)al(NN * 4 * 4);
  unsigned* EMAX = (unsigned*)al(NN * 4 * 4);
  float* DEN = (float*)al(NN * 4 * 4);
  float* EBUF= (float*)al(EE * 4 * 4);
  float* IDEN= (float*)al(NN * 64 * 4);
  float* ST  = (float*)al(256 * 4);       // column sums / sumsq

  auto blks = [](long n) { return dim3((unsigned)((n + 255) / 256)); };
  const dim3 T(256);
  const dim3 GMM((unsigned)(NN / 16));    // 6250 row tiles

  // ---- prep: bf16 copies and transposed weights ----
  k_f32_to_bf16<<<blks(NN * 128), T, 0, stream>>>(X16, x, NN * 128);
  k_transpose_bf16<<<blks(128 * 128), T, 0, stream>>>(WL1T, Wl1, 128, 128);
  k_transpose_bf16<<<blks(128 * 128), T, 0, stream>>>(WR1T, Wr1, 128, 128);
  k_transpose_bf16<<<blks(128 * 512), T, 0, stream>>>(WGT, Wg, 128, 512);
  k_transpose_bf16<<<blks(128 * 64), T, 0, stream>>>(WL3T, Wl3, 128, 64);
  k_transpose_bf16<<<blks(128 * 64), T, 0, stream>>>(WR3T, Wr3, 128, 64);
  k_transpose_bf16<<<blks(128 * 64), T, 0, stream>>>(WSKT, Wsk, 128, 64);
  k_transpose_bf16<<<blks(64 * 64), T, 0, stream>>>(WC1T, Wc1, 64, 64);
  k_transpose_bf16<<<blks(64 * 32), T, 0, stream>>>(WC2T, Wc2, 64, 32);

  // ---- degree (shared by both SAGE layers) ----
  k_zero_f32<<<blks(NN), T, 0, stream>>>(DEG, NN);
  k_deg<<<blks(EE), T, 0, stream>>>(DEG, dstI, EE);

  // ---- skip connection: identity = x @ Wsk + bsk ----
  k_gemm<<<GMM, T, 0, stream>>>(X16, nullptr, WSKT, nullptr, bsk, IDEN, nullptr, 128, 64, 0);

  // ---- SAGE1 ----
  k_zero_f32<<<blks(NN * 128), T, 0, stream>>>(A32, NN * 128);
  k_scatter_add_f32<<<blks(EE * 128), T, 0, stream>>>(A32, x, srcI, dstI, EE * 128, 128);
  k_mean_bf16<<<blks(NN * 128), T, 0, stream>>>(M16, A32, DEG, NN * 128, 128);
  k_gemm<<<GMM, T, 0, stream>>>(M16, X16, WL1T, WR1T, b1, A32, nullptr, 128, 128, 0);
  k_zero_f32<<<blks(256), T, 0, stream>>>(ST, 256);
  k_colstats<<<dim3(1024), T, 0, stream>>>(A32, ST, NN, 128);
  k_bn_apply<<<blks(NN * 128), T, 0, stream>>>(A32, ST, g1, be1, nullptr, H1B,
                                               NN * 128, 128, 1.f / NN, 1);

  // ---- GAT ----
  k_gemm<<<GMM, T, 0, stream>>>(H1B, nullptr, WGT, nullptr, nullptr, nullptr, XH, 128, 512, 0);
  k_attn_scores<<<blks(NN * 4), T, 0, stream>>>(AS, AD, XH, atts, attd, NN * 4);
  k_fill_u32<<<blks(NN * 4), T, 0, stream>>>(EMAX, 0u, NN * 4);
  k_edge_scores<<<blks(EE * 4), T, 0, stream>>>(EBUF, EMAX, AS, AD, srcI, dstI, EE * 4);
  k_zero_f32<<<blks(NN * 4), T, 0, stream>>>(DEN, NN * 4);
  k_edge_exp<<<blks(EE * 4), T, 0, stream>>>(EBUF, DEN, EMAX, dstI, EE * 4);
  k_zero_f32<<<blks(NN * 128), T, 0, stream>>>(A32, NN * 128);
  k_gat_agg<<<blks(EE * 128), T, 0, stream>>>(A32, XH, EBUF, DEN, srcI, dstI, EE * 128);
  k_axpb<<<blks(NN * 128), T, 0, stream>>>(A32, 0.25f, bg, NN * 128, 128); // mean heads + bias
  k_zero_f32<<<blks(256), T, 0, stream>>>(ST, 256);
  k_colstats<<<dim3(1024), T, 0, stream>>>(A32, ST, NN, 128);
  k_bn_apply<<<blks(NN * 128), T, 0, stream>>>(A32, ST, g2, be2, nullptr, H2B,
                                               NN * 128, 128, 1.f / NN, 1);

  // ---- SAGE3 ----
  k_zero_f32<<<blks(NN * 128), T, 0, stream>>>(A32, NN * 128);
  k_scatter_add_bf16<<<blks(EE * 128), T, 0, stream>>>(A32, H2B, srcI, dstI, EE * 128, 128);
  k_mean_bf16<<<blks(NN * 128), T, 0, stream>>>(M16, A32, DEG, NN * 128, 128);
  k_gemm<<<GMM, T, 0, stream>>>(M16, H2B, WL3T, WR3T, b3, B32, nullptr, 128, 64, 0);
  k_zero_f32<<<blks(256), T, 0, stream>>>(ST, 256);
  k_colstats<<<dim3(1024), T, 0, stream>>>(B32, ST, NN, 64);
  k_bn_apply<<<blks(NN * 64), T, 0, stream>>>(B32, ST, g3, be3, H3F, nullptr,
                                              NN * 64, 64, 1.f / NN, 1);

  // ---- emb = h3 + identity ; classifier ----
  k_add_emb<<<blks(NN * 64), T, 0, stream>>>(EMB, H3F, IDEN, NN * 64);
  k_gemm<<<GMM, T, 0, stream>>>(EMB, nullptr, WC1T, nullptr, bc1, B32, nullptr, 64, 64, 0);
  k_zero_f32<<<blks(256), T, 0, stream>>>(ST, 256);
  k_colstats<<<dim3(1024), T, 0, stream>>>(B32, ST, NN, 64);
  k_bn_apply<<<blks(NN * 64), T, 0, stream>>>(B32, ST, gc, bec, nullptr, Z1B,
                                              NN * 64, 64, 1.f / NN, 1);
  k_gemm<<<GMM, T, 0, stream>>>(Z1B, nullptr, WC2T, nullptr, bc2, nullptr, Z2B, 64, 32, 1);
  k_head<<<blks(NN), T, 0, stream>>>(out, Z2B, Wc3, bc3, NN);
}
